// LPEBlock_74586402062456
// MI455X (gfx1250) — matrix operand order repaired and checked by reading, not compiled, not run on hardware
//
#include <hip/hip_runtime.h>
#include <hip/hip_bf16.h>
#include <math.h>

// Problem constants
#define BB   4
#define NN   16384
#define DD   512
#define CC   2048          // clusters per batch
#define NSEG (BB * CC)     // 8192
#define NPTS (BB * NN)     // 65536
#define BND  ((size_t)NPTS * DD)   // 33554432 elements per output tensor
#define MPB  64            // points (M rows) per block

typedef __attribute__((ext_vector_type(16))) __bf16 v16bf;
typedef __attribute__((ext_vector_type(8)))  float  v8f;

union Frag16 { v16bf v; uint4 q[2]; };

static __device__ __forceinline__ unsigned short f2bf(float f) {
    union { float f; unsigned u; } v; v.f = f;
    unsigned u = v.u;
    u += 0x7FFFu + ((u >> 16) & 1u);     // round-to-nearest-even
    return (unsigned short)(u >> 16);
}

// ---------------- Phase 0: zero accumulators ----------------
__global__ void lpe_zero_k(float* p, int n) {
    int i = blockIdx.x * blockDim.x + threadIdx.x;
    if (i < n) p[i] = 0.0f;
}

// ---------------- Phase 1: segment sums of pos + counts ----------------
__global__ void lpe_accum_pos_k(const float* __restrict__ pos,
                                const int* __restrict__ cid,
                                float* __restrict__ sumP, float* __restrict__ cnt) {
    int p = blockIdx.x * blockDim.x + threadIdx.x;
    if (p >= NPTS) return;
    int b  = p >> 14;                     // p / N
    int sg = cid[p] + (b << 11);          // + b*C
    atomicAdd(&sumP[3 * sg + 0], pos[3 * p + 0]);
    atomicAdd(&sumP[3 * sg + 1], pos[3 * p + 1]);
    atomicAdd(&sumP[3 * sg + 2], pos[3 * p + 2]);
    atomicAdd(&cnt[sg], 1.0f);
}

// cog/avg = sums / clip(count,1)
__global__ void lpe_finalize_k(const float* __restrict__ sums,
                               const float* __restrict__ cnt,
                               float* __restrict__ dst) {
    int i = blockIdx.x * blockDim.x + threadIdx.x;
    if (i >= NSEG) return;
    float c = fmaxf(cnt[i], 1.0f);
    dst[3 * i + 0] = sums[3 * i + 0] / c;
    dst[3 * i + 1] = sums[3 * i + 1] / c;
    dst[3 * i + 2] = sums[3 * i + 2] / c;
}

// ---------------- Phase 2: segment sums of local_p ----------------
__global__ void lpe_accum_lp_k(const float* __restrict__ pos,
                               const int* __restrict__ cid,
                               const float* __restrict__ cog,
                               float* __restrict__ sumLP) {
    int p = blockIdx.x * blockDim.x + threadIdx.x;
    if (p >= NPTS) return;
    int b  = p >> 14;
    int sg = cid[p] + (b << 11);
    atomicAdd(&sumLP[3 * sg + 0], pos[3 * p + 0] - cog[3 * sg + 0]);
    atomicAdd(&sumLP[3 * sg + 1], pos[3 * p + 1] - cog[3 * sg + 1]);
    atomicAdd(&sumLP[3 * sg + 2], pos[3 * p + 2] - cog[3 * sg + 2]);
}

// ---------------- Phase 3: pack w2/w4 -> bf16 in B-fragment order ----------------
// Fragment linear layout: idx = ((ktile*32 + ntile)*32 + lane)*16 + e
// with  n = ntile*16 + (lane&15),  k = ktile*32 + (lane>>4)*16 + e   (B: 32x16 per tile)
__global__ void lpe_pack_w_k(const float* __restrict__ w2,
                             const float* __restrict__ w4,
                             unsigned short* __restrict__ w2b,
                             unsigned short* __restrict__ w4b) {
    int idx = blockIdx.x * blockDim.x + threadIdx.x;   // 2 * 512*512 elements
    if (idx >= 2 * DD * DD) return;
    int which = idx >> 18;          // 262144 per matrix
    int fi    = idx & 0x3FFFF;
    int e     = fi & 15;
    int lane  = (fi >> 4) & 31;
    int tile  = fi >> 9;
    int ntile = tile & 31;
    int ktile = tile >> 5;
    int k = (ktile << 5) + ((lane >> 4) << 4) + e;
    int n = (ntile << 4) + (lane & 15);
    const float* src = which ? w4 : w2;
    unsigned short* dst = which ? w4b : w2b;
    dst[fi] = f2bf(src[k * DD + n]);
}

// ---------------- Phase 4: fused layer1 + WMMA GEMM + epilogue ----------------
// One block = 64 points x full D=512 output, for both MLPs.
// Each wave owns an N-tile column and all 4 M-tiles -> one B-fragment load feeds
// 4 independent WMMA accumulator chains (fills the 4 co-exec hazard slots).
__global__ __launch_bounds__(256)
void lpe_main_k(const float* __restrict__ pos,
                const float* __restrict__ feat,
                const int*   __restrict__ cid,
                const float* __restrict__ w1, const float* __restrict__ b1,
                const float* __restrict__ b2,
                const float* __restrict__ w3, const float* __restrict__ b3,
                const float* __restrict__ b4,
                const float* __restrict__ cog, const float* __restrict__ avg,
                const unsigned short* __restrict__ w2b,
                const unsigned short* __restrict__ w4b,
                float* __restrict__ out) {
    extern __shared__ __align__(16) unsigned short hid[];  // MPB*DD bf16 = 64 KB dynamic
    __shared__ float xbuf[MPB][8];                         // lp.xyz, nrm, avg.xyz, pad

    const int tid  = threadIdx.x;
    const int lane = tid & 31;
    const int wave = tid >> 5;
    const int blockbase = blockIdx.x * MPB;

    // ---- load per-point geometry ----
    if (tid < MPB) {
        int p  = blockbase + tid;
        int b  = p >> 14;
        int sg = cid[p] + (b << 11);
        float lx = pos[3 * p + 0] - cog[3 * sg + 0];
        float ly = pos[3 * p + 1] - cog[3 * sg + 1];
        float lz = pos[3 * p + 2] - cog[3 * sg + 2];
        xbuf[tid][0] = lx; xbuf[tid][1] = ly; xbuf[tid][2] = lz;
        xbuf[tid][3] = sqrtf(lx * lx + ly * ly + lz * lz);
        xbuf[tid][4] = avg[3 * sg + 0];
        xbuf[tid][5] = avg[3 * sg + 1];
        xbuf[tid][6] = avg[3 * sg + 2];
        xbuf[tid][7] = 0.0f;
    }
    __syncthreads();

    for (int which = 0; which < 2; ++which) {
        // ---- layer 1: hidden[64][512] = relu(x @ wA + bA), stored bf16 in LDS ----
        for (int idx = tid; idx < MPB * DD; idx += 256) {
            int m = idx >> 9, j = idx & 511;
            float s;
            if (which == 0) {
                s = b1[j]
                  + xbuf[m][0] * w1[0 * DD + j] + xbuf[m][1] * w1[1 * DD + j]
                  + xbuf[m][2] * w1[2 * DD + j] + xbuf[m][3] * w1[3 * DD + j];
            } else {
                s = b3[j]
                  + xbuf[m][4] * w3[0 * DD + j] + xbuf[m][5] * w3[1 * DD + j]
                  + xbuf[m][6] * w3[2 * DD + j] + xbuf[m][0] * w3[3 * DD + j]
                  + xbuf[m][1] * w3[4 * DD + j] + xbuf[m][2] * w3[5 * DD + j];
            }
            hid[idx] = f2bf(fmaxf(s, 0.0f));
        }
        __syncthreads();

        // ---- layer 2: [64x512] @ [512x512] via v_wmma_f32_16x16x32_bf16 ----
        const unsigned short* wb  = which ? w4b : w2b;
        const float*          bia = which ? b4  : b2;
        const size_t whichOff = which ? BND : 0;

        for (int ntile = wave; ntile < 32; ntile += 8) {   // uniform per wave: 4 iters
            v8f acc0 = {}, acc1 = {}, acc2 = {}, acc3 = {};
            const int mrow = (lane & 15) * DD;             // M-tile 0 row base
            #pragma unroll 4
            for (int ks = 0; ks < 16; ++ks) {
                // B fragment: two contiguous b128 per lane from swizzled weights
                Frag16 bf;
                const uint4* bq = (const uint4*)wb + ((((ks << 5) + ntile) << 5) + lane) * 2;
                bf.q[0] = bq[0];
                bf.q[1] = bq[1];
                // A fragments for the 4 M-tiles: two b128 LDS loads each
                int kb = (ks << 5) + ((lane >> 4) << 4);
                Frag16 a0, a1, a2, a3;
                a0.q[0] = *(const uint4*)&hid[mrow + kb];
                a0.q[1] = *(const uint4*)&hid[mrow + kb + 8];
                a1.q[0] = *(const uint4*)&hid[mrow + 16 * DD + kb];
                a1.q[1] = *(const uint4*)&hid[mrow + 16 * DD + kb + 8];
                a2.q[0] = *(const uint4*)&hid[mrow + 32 * DD + kb];
                a2.q[1] = *(const uint4*)&hid[mrow + 32 * DD + kb + 8];
                a3.q[0] = *(const uint4*)&hid[mrow + 48 * DD + kb];
                a3.q[1] = *(const uint4*)&hid[mrow + 48 * DD + kb + 8];
                acc0 = __builtin_amdgcn_wmma_f32_16x16x32_bf16(
                    false, a0.v, false, bf.v, (short)0, acc0, false, false);
                acc1 = __builtin_amdgcn_wmma_f32_16x16x32_bf16(
                    false, a1.v, false, bf.v, (short)0, acc1, false, false);
                acc2 = __builtin_amdgcn_wmma_f32_16x16x32_bf16(
                    false, a2.v, false, bf.v, (short)0, acc2, false, false);
                acc3 = __builtin_amdgcn_wmma_f32_16x16x32_bf16(
                    false, a3.v, false, bf.v, (short)0, acc3, false, false);
            }
            // ---- epilogue: + bias + feat, store f32 ----
            int col = (ntile << 4) + (lane & 15);
            float bv = bia[col];
            int mhalf = (lane >> 4) << 3;                  // D layout: m = half*8 + g
            #pragma unroll
            for (int g = 0; g < 8; ++g) {
                size_t o0 = (size_t)(blockbase +      mhalf + g) * DD + col;
                size_t o1 = (size_t)(blockbase + 16 + mhalf + g) * DD + col;
                size_t o2 = (size_t)(blockbase + 32 + mhalf + g) * DD + col;
                size_t o3 = (size_t)(blockbase + 48 + mhalf + g) * DD + col;
                out[whichOff + o0] = acc0[g] + bv + feat[o0];
                out[whichOff + o1] = acc1[g] + bv + feat[o1];
                out[whichOff + o2] = acc2[g] + bv + feat[o2];
                out[whichOff + o3] = acc3[g] + bv + feat[o3];
            }
        }
        __syncthreads();   // all waves done reading hid before next 'which' overwrites
    }
}

// ---------------- launch ----------------
extern "C" void kernel_launch(void* const* d_in, const int* in_sizes, int n_in,
                              void* d_out, int out_size, void* d_ws, size_t ws_size,
                              hipStream_t stream) {
    const float* pos  = (const float*)d_in[0];
    const float* feat = (const float*)d_in[1];
    const int*   cid  = (const int*)  d_in[2];
    const float* w1   = (const float*)d_in[3];
    const float* b1   = (const float*)d_in[4];
    const float* w2   = (const float*)d_in[5];
    const float* b2   = (const float*)d_in[6];
    const float* w3   = (const float*)d_in[7];
    const float* b3   = (const float*)d_in[8];
    const float* w4   = (const float*)d_in[9];
    const float* b4   = (const float*)d_in[10];
    float* out = (float*)d_out;

    // workspace layout
    float* ws    = (float*)d_ws;
    float* sumP  = ws;                   // 3*NSEG
    float* cnt   = sumP  + 3 * NSEG;     // NSEG
    float* sumLP = cnt   + NSEG;         // 3*NSEG
    float* cog   = sumLP + 3 * NSEG;     // 3*NSEG
    float* avg   = cog   + 3 * NSEG;     // 3*NSEG
    unsigned short* w2b = (unsigned short*)(avg + 3 * NSEG);   // 512*512 bf16
    unsigned short* w4b = w2b + DD * DD;

    // 0) zero accumulators (sumP, cnt, sumLP are contiguous: NSEG*7 floats)
    {
        int n = NSEG * 7;
        lpe_zero_k<<<(n + 255) / 256, 256, 0, stream>>>(sumP, n);
    }
    // 1) pos segment sums + counts -> cog
    lpe_accum_pos_k<<<NPTS / 256, 256, 0, stream>>>(pos, cid, sumP, cnt);
    lpe_finalize_k<<<NSEG / 256, 256, 0, stream>>>(sumP, cnt, cog);
    // 2) local_p segment sums -> avg
    lpe_accum_lp_k<<<NPTS / 256, 256, 0, stream>>>(pos, cid, cog, sumLP);
    lpe_finalize_k<<<NSEG / 256, 256, 0, stream>>>(sumLP, cnt, avg);
    // 3) pack weights to bf16 fragment order
    lpe_pack_w_k<<<(2 * DD * DD) / 256, 256, 0, stream>>>(w2, w4, w2b, w4b);
    // 4) fused MLPs with WMMA (64 KB dynamic LDS for the hidden activations)
    lpe_main_k<<<NPTS / MPB, 256, MPB * DD * sizeof(unsigned short), stream>>>(
        pos, feat, cid, w1, b1, b2, w3, b3, b4, cog, avg, w2b, w4b, out);
}
